// FP8Linear_11416023073336
// MI455X (gfx1250) — 8-bit matrix operand order repaired and checked by reading, compiled, not run on hardware
//
#include <hip/hip_runtime.h>

typedef __attribute__((ext_vector_type(16))) int   v16i;
typedef __attribute__((ext_vector_type(4)))  int   vi4;
typedef __attribute__((ext_vector_type(2)))  int   vi2;
typedef __attribute__((ext_vector_type(8)))  float v8f;
typedef __attribute__((ext_vector_type(4)))  float vf4;

typedef unsigned int u32x4 __attribute__((ext_vector_type(4)));
typedef int          i32x4v __attribute__((ext_vector_type(4)));
typedef int          i32x8v __attribute__((ext_vector_type(8)));

#define FP8_MAX 448.0f

#define LDSROW       144
#define A_TILE_ROWS  256
#define B_TILE_ROWS  128
#define A_TILE_BYTES (A_TILE_ROWS * LDSROW)
#define B_TILE_BYTES (B_TILE_ROWS * LDSROW)
#define BUF_BYTES    (A_TILE_BYTES + B_TILE_BYTES)
#define LDS_TOTAL    (2 * BUF_BYTES)

#if __has_builtin(__builtin_amdgcn_tensor_load_to_lds)
#define USE_TDM 1
#else
#define USE_TDM 0
#endif

__device__ inline unsigned int sw_f32_to_e4m3(float f) {
    unsigned int u = __float_as_uint(f);
    unsigned int s = (u >> 31) << 7;
    unsigned int a = u & 0x7FFFFFFFu;
    if (a >= 0x7F800000u) return s | 0x7Fu;
    float af = __uint_as_float(a);
    if (af >= 464.0f) return s | 0x7Eu;
    if (af < 0.015625f) {
        float q = rintf(af * 512.0f);
        return s | (unsigned int)q;
    }
    unsigned int r = a + 0x0007FFFFu + ((a >> 20) & 1u);
    unsigned int e = (r >> 23) - 120u;
    unsigned int m = (r >> 20) & 7u;
    if (e > 15u) { e = 15u; m = 6u; }
    return s | (e << 3) | m;
}

__device__ inline unsigned int pack4_e4m3(float a, float b, float c, float d) {
#if __has_builtin(__builtin_amdgcn_cvt_pk_fp8_f32)
    int p = __builtin_amdgcn_cvt_pk_fp8_f32(a, b, 0, false);
    p = __builtin_amdgcn_cvt_pk_fp8_f32(c, d, p, true);
    return (unsigned int)p;
#else
    return sw_f32_to_e4m3(a) | (sw_f32_to_e4m3(b) << 8) |
           (sw_f32_to_e4m3(c) << 16) | (sw_f32_to_e4m3(d) << 24);
#endif
}

__global__ void __launch_bounds__(256)
quant_fp8_kernel(const float* __restrict__ src,
                 unsigned int* __restrict__ dstq,
                 float* __restrict__ scaleT,
                 int kbPerRow, int rows) {
    const int waveInBlk = threadIdx.x >> 5;
    const int lane      = threadIdx.x & 31;
    const long blkIdx   = (long)blockIdx.x * (blockDim.x >> 5) + waveInBlk;
    const long base     = blkIdx * 128 + lane * 4;

    vf4 v = __builtin_nontemporal_load((const vf4*)(src + base));
    float m = fmaxf(fmaxf(fabsf(v.x), fabsf(v.y)), fmaxf(fabsf(v.z), fabsf(v.w)));
    #pragma unroll
    for (int off = 16; off >= 1; off >>= 1)
        m = fmaxf(m, __shfl_xor(m, off, 32));

    const float inv = (m > 0.0f) ? (FP8_MAX / m) : 0.0f;
    dstq[blkIdx * 32 + lane] =
        pack4_e4m3(v.x * inv, v.y * inv, v.z * inv, v.w * inv);

    if (lane == 0) {
        const int row = (int)(blkIdx / kbPerRow);
        const int kb  = (int)(blkIdx % kbPerRow);
        scaleT[(long)kb * rows + row] = m * (1.0f / FP8_MAX);
    }
}

__device__ inline void tdm_stage_tile(unsigned lds_addr,
                                      const unsigned char* gptr,
                                      unsigned rows, unsigned strideBytes) {
#if USE_TDM
    const unsigned long long ga = (unsigned long long)(uintptr_t)gptr;
    u32x4 g0;
    g0[0] = 1u;
    g0[1] = lds_addr;
    g0[2] = (unsigned)ga;
    g0[3] = (unsigned)(ga >> 32) | (2u << 30);

    const unsigned d0 = strideBytes;
    i32x8v g1;
    g1[0] = (int)((1u << 20) |
                  (4u << 22) |
                  (3u << 25));
    g1[1] = (int)((d0 & 0xFFFFu) << 16);
    g1[2] = (int)((d0 >> 16) | ((rows & 0xFFFFu) << 16));
    g1[3] = (int)((rows >> 16) | (128u << 16));
    g1[4] = (int)rows;
    g1[5] = (int)strideBytes;
    g1[6] = 0;
    g1[7] = 0;

    const i32x4v z4 = {0, 0, 0, 0};
#if __clang_major__ >= 23
    const i32x8v z8 = {0, 0, 0, 0, 0, 0, 0, 0};
    __builtin_amdgcn_tensor_load_to_lds(g0, g1, z4, z4, z8, 0);
#else
    __builtin_amdgcn_tensor_load_to_lds(g0, g1, z4, z4, 0);
#endif
#endif
}

__device__ inline void coop_stage_tile(unsigned char* lds,
                                       const unsigned char* gptr,
                                       unsigned rows, unsigned strideBytes) {
    const unsigned chunks = rows * 8;
    for (unsigned c = threadIdx.x; c < chunks; c += 256) {
        const unsigned r = c >> 3, col = (c & 7) * 16;
        *(vi4*)(lds + r * LDSROW + col) =
            *(const vi4*)(gptr + (long)r * strideBytes + col);
    }
}

__global__ void __launch_bounds__(256)
fp8_gemm_kernel(const unsigned char* __restrict__ qx, const float* __restrict__ sxT,
                const unsigned char* __restrict__ qw, const float* __restrict__ swT,
                float* __restrict__ out, int M, int N, int K) {
    extern __shared__ unsigned char smem[];
    const unsigned ldsBase = (unsigned)(uintptr_t)(void*)smem;

    const int KB   = K >> 7;
    const int lane = threadIdx.x & 31;
    const int wv   = threadIdx.x >> 5;
    const int lm   = lane & 15;
    const int lh   = lane >> 4;

    const long tileN = (long)blockIdx.x * B_TILE_ROWS;
    const long wgM   = (long)blockIdx.y * A_TILE_ROWS;
    const long rowBase = wgM + wv * 32;

    const unsigned char* aG = qx + wgM * (long)K;
    const unsigned char* bG = qw + tileN * (long)K;
    const float* sxBase = sxT + rowBase + lh * 8;

    v8f acc0[8] = {}, acc1[8] = {};

#if USE_TDM
    if (wv == 0) {
        tdm_stage_tile(ldsBase, aG, A_TILE_ROWS, (unsigned)K);
        tdm_stage_tile(ldsBase + A_TILE_BYTES, bG, B_TILE_ROWS, (unsigned)K);
        __builtin_amdgcn_s_wait_tensorcnt(0);
    }
#else
    coop_stage_tile(smem, aG, A_TILE_ROWS, (unsigned)K);
    coop_stage_tile(smem + A_TILE_BYTES, bG, B_TILE_ROWS, (unsigned)K);
#endif
    __syncthreads();

    for (int kb = 0; kb < KB; ++kb) {
        const int cur = kb & 1;
        const unsigned long long ko = (unsigned long long)(kb + 1) * 128ull;

        if (kb + 1 < KB) {
#if USE_TDM
            if (wv == 0) {
                const unsigned nb = ldsBase + (unsigned)((cur ^ 1) * BUF_BYTES);
                tdm_stage_tile(nb, aG + ko, A_TILE_ROWS, (unsigned)K);
                tdm_stage_tile(nb + (unsigned)A_TILE_BYTES, bG + ko,
                               B_TILE_ROWS, (unsigned)K);
            }
#else
            unsigned char* nb = smem + (cur ^ 1) * BUF_BYTES;
            coop_stage_tile(nb, aG + ko, A_TILE_ROWS, (unsigned)K);
            coop_stage_tile(nb + A_TILE_BYTES, bG + ko, B_TILE_ROWS, (unsigned)K);
#endif
        }

        const unsigned char* As =
            smem + cur * BUF_BYTES + (wv * 32 + lm) * LDSROW + lh * 16;
        v16i a0, a1;
        #pragma unroll
        for (int j = 0; j < 8; ++j) {
            ((vi2*)&a0)[j] = *(const vi2*)(As + (j >> 1) * 32 + (j & 1) * 8);
            ((vi2*)&a1)[j] = *(const vi2*)(As + 16 * LDSROW + (j >> 1) * 32 + (j & 1) * 8);
        }

        v8f sx0, sx1;
        const float* sp = sxBase + (long)kb * M;
        ((vf4*)&sx0)[0] = *(const vf4*)(sp);
        ((vf4*)&sx0)[1] = *(const vf4*)(sp + 4);
        ((vf4*)&sx1)[0] = *(const vf4*)(sp + 16);
        ((vf4*)&sx1)[1] = *(const vf4*)(sp + 20);

        const float* swp = swT + (long)kb * N + tileN + lm;
        const unsigned char* Bs =
            smem + cur * BUF_BYTES + A_TILE_BYTES + lm * LDSROW + lh * 16;

        #pragma unroll
        for (int nt = 0; nt < 8; ++nt) {
            const unsigned char* bp = Bs + nt * 16 * LDSROW;
            v16i b;
            #pragma unroll
            for (int g = 0; g < 4; ++g)
                ((vi4*)&b)[g] = *(const vi4*)(bp + g * 32);

            const float sw = swp[nt * 16];
            v8f c = {};
            v8f d0 = __builtin_amdgcn_wmma_f32_16x16x128_fp8_fp8(
                a0, b, (short)0, c, false, false);
            #pragma unroll
            for (int i = 0; i < 8; ++i)
                acc0[nt][i] += d0[i] * (sx0[i] * sw);

            v8f d1 = __builtin_amdgcn_wmma_f32_16x16x128_fp8_fp8(
                a1, b, (short)0, c, false, false);
            #pragma unroll
            for (int i = 0; i < 8; ++i)
                acc1[nt][i] += d1[i] * (sx1[i] * sw);
        }

#if USE_TDM
        if (wv == 0) __builtin_amdgcn_s_wait_tensorcnt(0);
#endif
        __syncthreads();
    }

    const long orow0 = rowBase + lh * 8;
    #pragma unroll
    for (int nt = 0; nt < 8; ++nt) {
        const long ocol = tileN + nt * 16 + lm;
        #pragma unroll
        for (int i = 0; i < 8; ++i) {
            __builtin_nontemporal_store(acc0[nt][i],
                out + (orow0 + i) * (long)N + ocol);
            __builtin_nontemporal_store(acc1[nt][i],
                out + (orow0 + 16 + i) * (long)N + ocol);
        }
    }
}

extern "C" void kernel_launch(void* const* d_in, const int* in_sizes, int n_in,
                              void* d_out, int out_size, void* d_ws, size_t ws_size,
                              hipStream_t stream) {
    const float* x = (const float*)d_in[0];
    const float* w = (const float*)d_in[1];
    float* out = (float*)d_out;

    const int K = 2048;
    const int M = in_sizes[0] / K;
    const int N = in_sizes[1] / K;
    const int KB = K / 128;

    unsigned char* qx = (unsigned char*)d_ws;
    unsigned char* qw = qx + (size_t)M * K;
    float* sxT = (float*)(qw + (size_t)N * K);
    float* swT = sxT + (size_t)KB * M;

    {
        const long xBlocks = (long)M * K / 128;
        quant_fp8_kernel<<<dim3((unsigned)(xBlocks / 8)), dim3(256), 0, stream>>>(
            x, (unsigned int*)qx, sxT, KB, M);
        const long wBlocks = (long)N * K / 128;
        quant_fp8_kernel<<<dim3((unsigned)(wBlocks / 8)), dim3(256), 0, stream>>>(
            w, (unsigned int*)qw, swT, KB, N);
    }

    {
        dim3 grid((unsigned)(N / B_TILE_ROWS), (unsigned)(M / A_TILE_ROWS));
        fp8_gemm_kernel<<<grid, dim3(256), LDS_TOTAL, stream>>>(
            qx, sxT, qw, swT, out, M, N, K);
    }
}
